// ResidualAttentionalGNN4_66975720013949
// MI455X (gfx1250) — compile-verified, matrix operand order checked by reading, not run on hardware
//
#include <hip/hip_runtime.h>
#include <math.h>

// ---- problem constants (match reference) ----
#define NN    61440      // B*F nodes
#define FD    240        // F_DIM
#define HD    240        // HID
#define BG    256        // batch (graphs)
#define EE    491520     // edges
#define LL    3
#define DEG_  8
#define TRIU_ 28680
#define IN1_  29400
#define EPS_  1e-5f

typedef __bf16       v16bf __attribute__((ext_vector_type(16)));
typedef float        v8f   __attribute__((ext_vector_type(8)));
typedef unsigned int v8u   __attribute__((ext_vector_type(8)));
typedef unsigned int v4u   __attribute__((ext_vector_type(4)));

static __device__ __forceinline__ unsigned bf16b(float f) {   // RNE f32 -> bf16 bits
  unsigned u = __builtin_bit_cast(unsigned, f);
  return (u + 0x7FFFu + ((u >> 16) & 1u)) >> 16;
}

// ---------------------------------------------------------------------------
// Staging: A[M,K] f32 -> row-major bf16, K zero-padded to Kpad (mult of 32),
// stored as dwords of (k even | k odd) pairs.
// ---------------------------------------------------------------------------
__global__ void k_pack_a(const float* __restrict__ A, unsigned* __restrict__ Ap,
                         int M, int K, int Kpad)
{
  const int half = Kpad >> 1;
  int idx = blockIdx.x * blockDim.x + threadIdx.x;
  if (idx >= M * half) return;
  int m = idx / half, kk = idx - m * half;
  int k0 = kk * 2;
  float lo = (k0     < K) ? A[(size_t)m * K + k0]     : 0.0f;
  float hi = (k0 + 1 < K) ? A[(size_t)m * K + k0 + 1] : 0.0f;
  Ap[idx] = bf16b(lo) | (bf16b(hi) << 16);
}

// W[K,Nout] f32 -> Wp[(Kpad/2), Nout] dwords: Wp[kk*N+n] = bf16 W[2kk,n] | bf16 W[2kk+1,n]<<16
__global__ void k_pack_w(const float* __restrict__ W, unsigned* __restrict__ Wp,
                         int K, int Nout, int Kpad)
{
  const int half = Kpad >> 1;
  int idx = blockIdx.x * blockDim.x + threadIdx.x;
  if (idx >= half * Nout) return;
  int kk = idx / Nout, n = idx - kk * Nout;
  int k0 = kk * 2;
  float lo = (k0     < K) ? W[(size_t)k0       * Nout + n] : 0.0f;
  float hi = (k0 + 1 < K) ? W[(size_t)(k0 + 1) * Nout + n] : 0.0f;
  Wp[idx] = bf16b(lo) | (bf16b(hi) << 16);
}

// ---------------------------------------------------------------------------
// GEMM: C[M,NOUT] = A @ B + bias, bf16 WMMA / f32 acc.
// NOUT is a template constant -> all strided B loads use immediate offsets.
// One wave computes a 16x64 strip (4 tiles, A fragment reused 4x); block = 128
// (4 waves = 256 cols); grid = (M/16, ceil(NOUT/256)). Guard-free hot loop:
// OOB rows/cols are clamped or read slab garbage; their outputs never stored.
// ---------------------------------------------------------------------------
template<int NOUT>
__global__ __launch_bounds__(128) void k_gemm_wmma(
    const unsigned* __restrict__ Ap, const unsigned* __restrict__ Bp,
    const float* __restrict__ bias, float* __restrict__ C,
    int M, int Kpad, int ep)
{
  const int lane    = threadIdx.x & 31;
  const int wave    = threadIdx.x >> 5;
  const int rowBase = blockIdx.x * 16;
  const int colBase = (blockIdx.y * 4 + wave) * 64;
  if (colBase >= NOUT) return;                  // wave-uniform, EXEC stays full
  const int hh  = lane >> 4;                    // wave half -> K sub-block
  const int lid = lane & 15;
  int m = rowBase + lid; if (m >= M) m = M - 1;         // clamp, never stored
  int nb = colBase + lid; if (nb > NOUT - 1) nb = NOUT - 1;
  const int half = Kpad >> 1;

  const unsigned* ap = Ap + (size_t)m * half + 4 * hh;
  const unsigned* bp = Bp + (size_t)(4 * hh) * NOUT + nb;

  v8f acc[4] = {};
  const int chunks = Kpad >> 5;
  for (int c = 0; c < chunks; ++c) {
    v4u al = *(const v4u*)(ap);        // k = k0+8h   .. +7  (e0..7)
    v4u ah = *(const v4u*)(ap + 8);    // k = k0+16+8h.. +7  (e8..15)
    ap += 16;
    v8u aw;
    aw[0] = al[0]; aw[1] = al[1]; aw[2] = al[2]; aw[3] = al[3];
    aw[4] = ah[0]; aw[5] = ah[1]; aw[6] = ah[2]; aw[7] = ah[3];
    v16bf a = __builtin_bit_cast(v16bf, aw);
#pragma unroll
    for (int t = 0; t < 4; ++t) {
      const unsigned* bt = bp + 16 * t;          // +16 cols, immediate offset
      v8u bw;
      bw[0] = bt[0];
      bw[1] = bt[1 * NOUT];
      bw[2] = bt[2 * NOUT];
      bw[3] = bt[3 * NOUT];
      bw[4] = bt[8 * NOUT];
      bw[5] = bt[9 * NOUT];
      bw[6] = bt[10 * NOUT];
      bw[7] = bt[11 * NOUT];
      v16bf b = __builtin_bit_cast(v16bf, bw);
      acc[t] = __builtin_amdgcn_wmma_f32_16x16x32_bf16(
                   false, a, false, b, (short)0, acc[t], false, false);
    }
    bp += 16 * NOUT;
  }
#pragma unroll
  for (int t = 0; t < 4; ++t) {
    const int nst = colBase + 16 * t + lid;
    if (nst < NOUT) {
      const float bv = bias[nst];
#pragma unroll
      for (int r = 0; r < 8; ++r) {
        const int mo = rowBase + r + 8 * hh;
        float v = acc[t][r] + bv;
        if (ep == 1) v = fmaxf(v, 0.0f);
        if (mo < M) C[(size_t)mo * NOUT + nst] = v;
      }
    }
  }
}

// ---------------------------------------------------------------------------
// Degree / edge-coefficient precompute (edge list is constant across layers)
// ---------------------------------------------------------------------------
__global__ void k_fill_deg(float* __restrict__ deg) {
  int i = blockIdx.x * blockDim.x + threadIdx.x;
  if (i < NN) deg[i] = 1.0f;                   // +1 self loop
}
__global__ void k_count_deg(const int* __restrict__ dst, float* __restrict__ deg) {
  int e = blockIdx.x * blockDim.x + threadIdx.x;
  if (e < EE) atomicAdd(&deg[dst[e]], 1.0f);
}
__global__ void k_coef(const int* __restrict__ src, const int* __restrict__ dst,
                       const float* __restrict__ deg, float* __restrict__ coef,
                       float* __restrict__ dinv) {
  int e = blockIdx.x * blockDim.x + threadIdx.x;
  if (e < EE) coef[e] = rsqrtf(deg[src[e]]) * rsqrtf(deg[dst[e]]);
  if (e < NN) dinv[e] = 1.0f / deg[e];
}

// ---------------------------------------------------------------------------
// GCN aggregation, one workgroup per graph. The graph's 240x240 output block
// (230 KB) lives in LDS (CDNA5 WGP has 320 KB). acc = xw/deg, then LDS-atomic
// scatter of the graph's 1920 contiguous edges, then X = tanh(acc).
// ---------------------------------------------------------------------------
__global__ __launch_bounds__(256) void k_agg_tanh(
    const float* __restrict__ xw, const float* __restrict__ dinv,
    const int* __restrict__ src, const int* __restrict__ dst,
    const float* __restrict__ coef, float* __restrict__ Xout)
{
  extern __shared__ float acc[];               // FD*HD floats = 230400 B
  const int g = blockIdx.x;
  const int nodeBase = g * FD;
  for (int idx = threadIdx.x; idx < FD * HD; idx += blockDim.x) {
    int i = idx / HD;
    acc[idx] = xw[(size_t)nodeBase * HD + idx] * dinv[nodeBase + i];
  }
  __syncthreads();
  const int eBase = g * FD * DEG_;
  for (int idx = threadIdx.x; idx < FD * DEG_ * HD; idx += blockDim.x) {
    int e = idx / HD;
    int f = idx - e * HD;
    int ee = eBase + e;
    int s = src[ee];
    int d = dst[ee] - nodeBase;
    atomicAdd(&acc[d * HD + f], xw[(size_t)s * HD + f] * coef[ee]);
  }
  __syncthreads();
  for (int idx = threadIdx.x; idx < FD * HD; idx += blockDim.x)
    Xout[(size_t)nodeBase * HD + idx] = tanhf(acc[idx]);
}

// Row softmax over 240 features, X *= softmax(scores)
__global__ __launch_bounds__(256) void k_softmax_scale(
    const float* __restrict__ sc, float* __restrict__ X)
{
  __shared__ float buf[256];
  const int r = blockIdx.x, t = threadIdx.x;
  float v = (t < HD) ? sc[(size_t)r * HD + t] : -3.4e38f;
  buf[t] = v; __syncthreads();
  for (int s = 128; s > 0; s >>= 1) { if (t < s) buf[t] = fmaxf(buf[t], buf[t + s]); __syncthreads(); }
  float mx = buf[0]; __syncthreads();
  float ex = (t < HD) ? __expf(v - mx) : 0.0f;
  buf[t] = ex; __syncthreads();
  for (int s = 128; s > 0; s >>= 1) { if (t < s) buf[t] += buf[t + s]; __syncthreads(); }
  float inv = 1.0f / buf[0];
  if (t < HD) X[(size_t)r * HD + t] *= ex * inv;
}

// Per-graph mean over 240 nodes -> H[:, lcol:lcol+240]
__global__ void k_graph_mean(const float* __restrict__ X, float* __restrict__ H, int lcol) {
  int idx = blockIdx.x * blockDim.x + threadIdx.x;
  if (idx >= BG * HD) return;
  int g = idx / HD, f = idx - g * HD;
  const float* p = X + (size_t)g * FD * HD + f;
  float s = 0.0f;
  for (int i = 0; i < FD; ++i) s += p[(size_t)i * HD];
  H[(size_t)g * (HD * LL) + lcol + f] = s * (1.0f / FD);
}

// slog[g] = dot(Z[g,:240], Wa) + ba
__global__ __launch_bounds__(256) void k_gla_logit(
    const float* __restrict__ Z, const float* __restrict__ Wa,
    const float* __restrict__ ba, float* __restrict__ slog)
{
  __shared__ float buf[256];
  int g = blockIdx.x, t = threadIdx.x;
  buf[t] = (t < HD) ? Z[(size_t)g * HD + t] * Wa[t] : 0.0f;
  __syncthreads();
  for (int s = 128; s > 0; s >>= 1) { if (t < s) buf[t] += buf[t + s]; __syncthreads(); }
  if (t == 0) slog[g] = buf[0] + ba[0];
}

// softmax over the 256 graphs (single block)
__global__ __launch_bounds__(256) void k_softmax_graphs(
    const float* __restrict__ s, float* __restrict__ w)
{
  __shared__ float buf[256];
  int t = threadIdx.x;
  float v = s[t];
  buf[t] = v; __syncthreads();
  for (int st = 128; st > 0; st >>= 1) { if (t < st) buf[t] = fmaxf(buf[t], buf[t + st]); __syncthreads(); }
  float mx = buf[0]; __syncthreads();
  float e = __expf(v - mx);
  buf[t] = e; __syncthreads();
  for (int st = 128; st > 0; st >>= 1) { if (t < st) buf[t] += buf[t + st]; __syncthreads(); }
  w[t] = e / buf[0];
}

// MLPin[g, TRIU+c] = BN(H[g,c] * wsm[g])
__global__ void k_bnh(const float* __restrict__ H, const float* __restrict__ wsm,
                      const float* __restrict__ g_, const float* __restrict__ b_,
                      const float* __restrict__ m_, const float* __restrict__ v_,
                      float* __restrict__ MLPin)
{
  int idx = blockIdx.x * blockDim.x + threadIdx.x;
  if (idx >= BG * HD * LL) return;
  int g = idx / (HD * LL), c = idx - g * (HD * LL);
  float val = H[idx] * wsm[g];
  val = (val - m_[c]) * rsqrtf(v_[c] + EPS_) * g_[c] + b_[c];
  MLPin[(size_t)g * IN1_ + TRIU_ + c] = val;
}

// MLPin[g, t] = BN(strict-upper-tri of x0 per graph)
__global__ void k_feat(const float* __restrict__ x0,
                       const float* __restrict__ g_, const float* __restrict__ b_,
                       const float* __restrict__ m_, const float* __restrict__ v_,
                       float* __restrict__ MLPin)
{
  int idx = blockIdx.x * blockDim.x + threadIdx.x;
  if (idx >= BG * FD * FD) return;
  int g = idx / (FD * FD);
  int rem = idx - g * FD * FD;
  int i = rem / FD, j = rem - i * FD;
  if (j <= i) return;
  int t = i * FD - (i * (i + 1)) / 2 + (j - i - 1);   // triu_indices(k=1) rank
  float val = x0[(size_t)(g * FD + i) * FD + j];
  val = (val - m_[t]) * rsqrtf(v_[t] + EPS_) * g_[t] + b_[t];
  MLPin[(size_t)g * IN1_ + t] = val;
}

// in-place BN (eval) + relu, channel = idx % C
__global__ void k_bnrelu(float* __restrict__ X,
                         const float* __restrict__ g_, const float* __restrict__ b_,
                         const float* __restrict__ m_, const float* __restrict__ v_,
                         int total, int C)
{
  int idx = blockIdx.x * blockDim.x + threadIdx.x;
  if (idx >= total) return;
  int c = idx % C;
  float val = (X[idx] - m_[c]) * rsqrtf(v_[c] + EPS_) * g_[c] + b_[c];
  X[idx] = val > 0.0f ? val : 0.0f;
}

// GEMM launch helper (NOUT compile-time for immediate-offset B loads)
template<int NOUT>
static void launch_gemm(const float* A, const float* W, const float* bias, float* C,
                        int M, int K, int ep, unsigned* Apack, unsigned* Wpack,
                        hipStream_t stream)
{
  const int Kpad = (K + 31) & ~31;
  const int half = Kpad >> 1;
  k_pack_a<<<((size_t)M * half + 255) / 256, 256, 0, stream>>>(A, Apack, M, K, Kpad);
  k_pack_w<<<((size_t)half * NOUT + 255) / 256, 256, 0, stream>>>(W, Wpack, K, NOUT, Kpad);
  dim3 grid((M + 15) / 16, (NOUT + 255) / 256);
  k_gemm_wmma<NOUT><<<grid, 128, 0, stream>>>(Apack, Wpack, bias, C, M, Kpad, ep);
}

// ---------------------------------------------------------------------------
extern "C" void kernel_launch(void* const* d_in, const int* in_sizes, int n_in,
                              void* d_out, int out_size, void* d_ws, size_t ws_size,
                              hipStream_t stream) {
  (void)in_sizes; (void)n_in; (void)out_size; (void)ws_size;
  const float* x0   = (const float*)d_in[0];
  const int*   src  = (const int*)d_in[1];
  const int*   dst  = src + EE;
  const float* convW[LL] = {(const float*)d_in[3], (const float*)d_in[5], (const float*)d_in[7]};
  const float* convB[LL] = {(const float*)d_in[4], (const float*)d_in[6], (const float*)d_in[8]};
  const float* attW [LL] = {(const float*)d_in[9],  (const float*)d_in[11], (const float*)d_in[13]};
  const float* attB [LL] = {(const float*)d_in[10], (const float*)d_in[12], (const float*)d_in[14]};
  const float* glaW = (const float*)d_in[15], *glaB = (const float*)d_in[16];
  const float* gaW  = (const float*)d_in[17], *gaB  = (const float*)d_in[18];
  const float* bnG  = (const float*)d_in[19], *bnB = (const float*)d_in[20];
  const float* bnM  = (const float*)d_in[21], *bnV = (const float*)d_in[22];
  const float* bhG  = (const float*)d_in[23], *bhB = (const float*)d_in[24];
  const float* bhM  = (const float*)d_in[25], *bhV = (const float*)d_in[26];
  const float* m0W = (const float*)d_in[27], *m0B = (const float*)d_in[28];
  const float* b1G = (const float*)d_in[29], *b1B = (const float*)d_in[30];
  const float* b1M = (const float*)d_in[31], *b1V = (const float*)d_in[32];
  const float* m2W = (const float*)d_in[33], *m2B = (const float*)d_in[34];
  const float* b3G = (const float*)d_in[35], *b3B = (const float*)d_in[36];
  const float* b3M = (const float*)d_in[37], *b3V = (const float*)d_in[38];
  const float* m4W = (const float*)d_in[39], *m4B = (const float*)d_in[40];
  const float* b5G = (const float*)d_in[41], *b5B = (const float*)d_in[42];
  const float* b5M = (const float*)d_in[43], *b5V = (const float*)d_in[44];
  const float* m6W = (const float*)d_in[45], *m6B = (const float*)d_in[46];

  // workspace layout
  char* ws = (char*)d_ws;
  size_t off = 0;
  auto take = [&](size_t bytes) { size_t o = off; off = (off + bytes + 255) & ~(size_t)255; return o; };
  float*    X     = (float*)(ws + take((size_t)NN * HD * 4));        // activations
  float*    T1    = (float*)(ws + take((size_t)NN * HD * 4));        // xw / scores; later MLP input
  unsigned* Apack = (unsigned*)(ws + take((size_t)NN * 128 * 4));    // bf16-pair A staging (max M*Kpad/2)
  unsigned* Wpack = (unsigned*)(ws + take(((size_t)14704 * 512 + 256) * 4)); // W staging (+1KB OOB slack)
  float*    deg   = (float*)(ws + take((size_t)NN * 4));
  float*    dinv  = (float*)(ws + take((size_t)NN * 4));
  float*    coef  = (float*)(ws + take((size_t)EE * 4));
  float*    H     = (float*)(ws + take((size_t)BG * HD * LL * 4));
  float*    Z     = (float*)(ws + take((size_t)BG * HD * 4));
  float*    slog  = (float*)(ws + take((size_t)BG * 4));
  float*    wsm   = (float*)(ws + take((size_t)BG * 4));
  float*    O1    = (float*)(ws + take((size_t)BG * 512 * 4));
  float*    O2    = (float*)(ws + take((size_t)BG * 256 * 4));
  float*    O3    = (float*)(ws + take((size_t)BG * 256 * 4));
  float*    MLPin = T1;   // T1 dead after layer loop; IN1*BG*4 < NN*HD*4

  const size_t ldsBytes = (size_t)FD * HD * 4;   // 230400 B < 320 KB WGP LDS
  hipFuncSetAttribute((const void*)k_agg_tanh,
                      hipFuncAttributeMaxDynamicSharedMemorySize, (int)ldsBytes);

  hipMemcpyAsync(X, x0, (size_t)NN * HD * 4, hipMemcpyDeviceToDevice, stream);

  k_fill_deg <<<(NN + 255) / 256, 256, 0, stream>>>(deg);
  k_count_deg<<<(EE + 255) / 256, 256, 0, stream>>>(dst, deg);
  k_coef     <<<(EE + 255) / 256, 256, 0, stream>>>(src, dst, deg, coef, dinv);

  for (int l = 0; l < LL; ++l) {
    launch_gemm<HD>(X, convW[l], convB[l], T1, NN, HD, 0, Apack, Wpack, stream);
    k_agg_tanh<<<BG, 256, ldsBytes, stream>>>(T1, dinv, src, dst, coef, X);
    launch_gemm<HD>(X, attW[l], attB[l], T1, NN, HD, 0, Apack, Wpack, stream);
    k_softmax_scale<<<NN, 256, 0, stream>>>(T1, X);
    k_graph_mean<<<(BG * HD + 255) / 256, 256, 0, stream>>>(X, H, l * HD);
  }

  // graph-level attention
  launch_gemm<HD>(H, glaW, glaB, Z, BG, HD * LL, 1 /*relu*/, Apack, Wpack, stream);
  k_gla_logit<<<BG, 256, 0, stream>>>(Z, gaW, gaB, slog);
  k_softmax_graphs<<<1, 256, 0, stream>>>(slog, wsm);

  // assemble MLP input: [feat(TRIU) | BN(h*w)(720)]
  k_feat<<<(BG * FD * FD + 255) / 256, 256, 0, stream>>>(x0, bnG, bnB, bnM, bnV, MLPin);
  k_bnh <<<(BG * HD * LL + 255) / 256, 256, 0, stream>>>(H, wsm, bhG, bhB, bhM, bhV, MLPin);

  // MLP head
  launch_gemm<512>(MLPin, m0W, m0B, O1, BG, IN1_, 0, Apack, Wpack, stream);
  k_bnrelu<<<(BG * 512 + 255) / 256, 256, 0, stream>>>(O1, b1G, b1B, b1M, b1V, BG * 512, 512);
  launch_gemm<256>(O1, m2W, m2B, O2, BG, 512, 0, Apack, Wpack, stream);
  k_bnrelu<<<(BG * 256 + 255) / 256, 256, 0, stream>>>(O2, b3G, b3B, b3M, b3V, BG * 256, 256);
  launch_gemm<256>(O2, m4W, m4B, O3, BG, 256, 0, Apack, Wpack, stream);
  k_bnrelu<<<(BG * 256 + 255) / 256, 256, 0, stream>>>(O3, b5G, b5B, b5M, b5V, BG * 256, 256);
  launch_gemm<2>(O3, m6W, m6B, (float*)d_out, BG, 256, 0, Apack, Wpack, stream);
}
